// NaiveCustomLSTM_18597208391680
// MI455X (gfx1250) — compile-verified, matrix operand order checked by reading, not run on hardware
//
#include <hip/hip_runtime.h>

// LSTM: B=64, S=2048, I=128, H=256 on gfx1250 (wave32, WMMA, TDM).
// Persistent kernel: 16 WGs x 256 threads. WG 'wg' owns h-columns
// [wg*16, wg*16+16) i.e. gate columns {g*256 + wg*16 + j}. Weight slices are
// DMA'd via the Tensor Data Mover into LDS once, repacked into registers as
// v_wmma_f32_16x16x32_f16 B-fragments, and held there for the entire
// 2048-step recurrence. h is exchanged via a double-buffered f16 global
// buffer (L2-resident) + grid barrier each step. x is pre-converted to f16
// and the next step's x fragments are prefetched across the grid barrier.
// h-fragments are pinned into distinct registers via an empty asm so all 16
// b128 loads issue up front and only one L2 round trip is exposed per step.

#define B_  64
#define S_  2048
#define I_  128
#define H_  256
#define NWG 16
#define NTH 256

typedef _Float16 v16h __attribute__((ext_vector_type(16)));
typedef _Float16 h2   __attribute__((ext_vector_type(2)));
typedef _Float16 h4   __attribute__((ext_vector_type(4)));
typedef _Float16 h8   __attribute__((ext_vector_type(8)));
typedef float    v8f  __attribute__((ext_vector_type(8)));
typedef float    f2   __attribute__((ext_vector_type(2)));
typedef float    f4   __attribute__((ext_vector_type(4)));
typedef unsigned int u32x4 __attribute__((ext_vector_type(4)));
typedef int          i32x4 __attribute__((ext_vector_type(4)));
typedef int          i32x8 __attribute__((ext_vector_type(8)));

#if __has_builtin(__builtin_amdgcn_tensor_load_to_lds)
#define HAVE_TDM 1
#else
#define HAVE_TDM 0
#endif

__device__ __forceinline__ float sigf(float x) {
    return 1.0f / (1.0f + __expf(-x));
}
__device__ __forceinline__ float tanh_fast(float x) {
    return 2.0f / (1.0f + __expf(-2.0f * x)) - 1.0f;
}

// K index within a 16x32 f16 A/B fragment (half = lane>>4, pair p, pos).
// ISA 7.12.2: VGPR0: lanes0-15 K=0,1 / lanes16-31 K=8,9; VGPR4: K=16,17/24,25.
__device__ __forceinline__ int kmap_pair_base(int half, int p) {
    return (p < 4) ? (8 * half + 2 * p) : (16 + 8 * half + 2 * (p - 4));
}
__device__ __forceinline__ int kmap_e(int half, int e) {
    return kmap_pair_base(half, e >> 1) + (e & 1);
}

#if HAVE_TDM
// Issue a TDM 2D tile load: tile_w f32 elems x tile_h rows, row stride
// row_stride f32 elems, from gptr into LDS byte offset lds_off.
// D# layout per CDNA5 ISA ch.8 (group0: count/lds/global/type, group1: dims).
__device__ __forceinline__ void tdm_load_2d_f32(const float* gptr, unsigned lds_off,
                                                unsigned tile_w, unsigned tile_h,
                                                unsigned row_stride) {
    unsigned long long ga = (unsigned long long)(uintptr_t)gptr;
    u32x4 g0;
    g0[0] = 1u;                                            // count=1 (valid D#)
    g0[1] = lds_off;                                       // lds_addr (bytes)
    g0[2] = (unsigned)(ga & 0xFFFFFFFFu);                  // global_addr[31:0]
    g0[3] = (unsigned)((ga >> 32) & 0x1FFFFFFu) | (2u << 30); // ga[56:32], type=2
    i32x8 g1;
    g1[0] = (int)(2u << 16);                               // data_size=4B
    g1[1] = (int)((tile_w & 0xFFFFu) << 16);               // tensor_dim0[15:0]
    g1[2] = (int)(((tile_w >> 16) & 0xFFFFu) | ((tile_h & 0xFFFFu) << 16));
    g1[3] = (int)(((tile_h >> 16) & 0xFFFFu) | ((tile_w & 0xFFFFu) << 16)); // tile_dim0
    g1[4] = (int)(tile_h & 0xFFFFu);                       // tile_dim1 (tile_dim2=0)
    g1[5] = (int)row_stride;                               // tensor_dim0_stride lo
    g1[6] = 0;
    g1[7] = 0;
    i32x4 z4 = {0, 0, 0, 0};
#if defined(__clang_major__) && __clang_major__ >= 23
    i32x8 z8 = {0, 0, 0, 0, 0, 0, 0, 0};
    __builtin_amdgcn_tensor_load_to_lds(g0, g1, z4, z4, z8, 0);
#else
    __builtin_amdgcn_tensor_load_to_lds(g0, g1, z4, z4, 0);
#endif
}
#endif

// Load the 4 A-fragments of x_t (rows brow.., K=128) into ax[0..3].
__device__ __forceinline__ void load_x_frags(const _Float16* __restrict__ x16,
                                             const float* __restrict__ x,
                                             size_t browS, int t, int half,
                                             v16h ax[4]) {
    if (x16) {
        const _Float16* xb = x16 + (browS + (size_t)t) * I_;
        #pragma unroll
        for (int kk = 0; kk < 4; ++kk) {
            v16h a;
            #pragma unroll
            for (int p = 0; p < 8; ++p) {
                int k0 = kk * 32 + kmap_pair_base(half, p);
                h2 v = *(const h2*)(xb + k0);
                a[2 * p] = v.x; a[2 * p + 1] = v.y;
            }
            ax[kk] = a;
        }
    } else {
        const float* xb = x + (browS + (size_t)t) * I_;
        #pragma unroll
        for (int kk = 0; kk < 4; ++kk) {
            v16h a;
            #pragma unroll
            for (int p = 0; p < 8; ++p) {
                int k0 = kk * 32 + kmap_pair_base(half, p);
                f2 v = *(const f2*)(xb + k0);
                a[2 * p] = (_Float16)v.x; a[2 * p + 1] = (_Float16)v.y;
            }
            ax[kk] = a;
        }
    }
}

__global__ void __launch_bounds__(NTH, 1) lstm_init_kernel(int* bar, _Float16* h0) {
    int i = blockIdx.x * blockDim.x + threadIdx.x;
    if (i == 0) *bar = 0;
    if (i < B_ * H_) h0[i] = (_Float16)0.0f;
}

// Convert x [B*S*I] f32 -> f16 once (8 elems/thread, b128 in / b128 out).
__global__ void __launch_bounds__(NTH, 1) xconv_kernel(const float* __restrict__ xin,
                                                       _Float16* __restrict__ xout) {
    size_t i = ((size_t)blockIdx.x * NTH + threadIdx.x) * 8;
    f4 a = *(const f4*)(xin + i);
    f4 b = *(const f4*)(xin + i + 4);
    h8 o;
    #pragma unroll
    for (int e = 0; e < 4; ++e) { o[e] = (_Float16)a[e]; o[e + 4] = (_Float16)b[e]; }
    *(h8*)(xout + i) = o;
}

__global__ void __launch_bounds__(NTH, 1) lstm_persistent_kernel(
    const float* __restrict__ x,
    const _Float16* __restrict__ x16,     // may be null -> f32 path
    const float* __restrict__ U0, const float* __restrict__ W0, const float* __restrict__ bb0,
    const float* __restrict__ U1, const float* __restrict__ W1, const float* __restrict__ bb1,
    const float* __restrict__ U2, const float* __restrict__ W2, const float* __restrict__ bb2,
    const float* __restrict__ U3, const float* __restrict__ W3, const float* __restrict__ bb3,
    float* __restrict__ out,
    _Float16* __restrict__ hbuf0, _Float16* __restrict__ hbuf1,
    int* __restrict__ bar)
{
    const int wg   = blockIdx.x;      // 0..15
    const int tid  = threadIdx.x;     // 0..255 (8 waves)
    const int wave = tid >> 5;
    const int lane = tid & 31;
    const int n    = lane & 15;       // tile column (B/C/D) / tile row (A)
    const int half = lane >> 4;
    const int mb   = wave & 3;        // batch block (16 rows)
    const int g0   = (wave >> 2) * 2; // gates handled: g0, g0+1
    const int hc0  = wg * 16;         // first h-column of this WG

    const float* Uarr[4] = {U0, U1, U2, U3};
    const float* Warr[4] = {W0, W1, W2, W3};
    const float* barr[4] = {bb0, bb1, bb2, bb3};

    __shared__ __attribute__((aligned(16))) float gLDS[4][4][16][16]; // 16 KB, also TDM stage
    __shared__ __attribute__((aligned(16))) float cLDS[B_][16];       // cell state, 4 KB

    for (int idx = tid; idx < B_ * 16; idx += NTH)
        (&cLDS[0][0])[idx] = 0.0f;

    // ---- Preload weight slices into registers as WMMA B-fragments (f16) ----
    v16h wfrag[2][8];
    v16h ufrag[2][4];
    float bias[2];
    bias[0] = barr[g0][hc0 + n];
    bias[1] = barr[g0 + 1][hc0 + n];

#if HAVE_TDM
    float* stage = &gLDS[0][0][0][0];                 // 256x16 f32 staging tile
    const unsigned stage_off = (unsigned)(uintptr_t)stage; // low 32 bits = LDS byte addr
    for (int g = 0; g < 4; ++g) {
        const int mine = ((wave >> 2) == (g >> 1));
        const int gg = g & 1;
        // W_g slice: rows k=0..255, cols hc0..hc0+15 (stride H) -> LDS via TDM
        __syncthreads();
        if (wave == 0) {
            tdm_load_2d_f32(Warr[g] + hc0, stage_off, 16u, (unsigned)H_, (unsigned)H_);
            __builtin_amdgcn_s_wait_tensorcnt(0);
        }
        __syncthreads();
        if (mine) {
            #pragma unroll
            for (int kk = 0; kk < 8; ++kk) {
                v16h f;
                #pragma unroll
                for (int e = 0; e < 16; ++e) {
                    int k = kk * 32 + kmap_e(half, e);
                    f[e] = (_Float16)stage[k * 16 + n];
                }
                wfrag[gg][kk] = f;
            }
        }
        __syncthreads();
        // U_g slice: rows k=0..127
        if (wave == 0) {
            tdm_load_2d_f32(Uarr[g] + hc0, stage_off, 16u, (unsigned)I_, (unsigned)H_);
            __builtin_amdgcn_s_wait_tensorcnt(0);
        }
        __syncthreads();
        if (mine) {
            #pragma unroll
            for (int kk = 0; kk < 4; ++kk) {
                v16h f;
                #pragma unroll
                for (int e = 0; e < 16; ++e) {
                    int k = kk * 32 + kmap_e(half, e);
                    f[e] = (_Float16)stage[k * 16 + n];
                }
                ufrag[gg][kk] = f;
            }
        }
    }
#else
    #pragma unroll
    for (int gg = 0; gg < 2; ++gg) {
        const int g = g0 + gg;
        const float* Wp = Warr[g];
        const float* Up = Uarr[g];
        #pragma unroll
        for (int kk = 0; kk < 8; ++kk) {
            v16h f;
            #pragma unroll
            for (int e = 0; e < 16; ++e)
                f[e] = (_Float16)Wp[(kk * 32 + kmap_e(half, e)) * H_ + hc0 + n];
            wfrag[gg][kk] = f;
        }
        #pragma unroll
        for (int kk = 0; kk < 4; ++kk) {
            v16h f;
            #pragma unroll
            for (int e = 0; e < 16; ++e)
                f[e] = (_Float16)Up[(kk * 32 + kmap_e(half, e)) * H_ + hc0 + n];
            ufrag[gg][kk] = f;
        }
    }
#endif

    __syncthreads();

    const int brow = mb * 16 + n;             // batch row for A-fragments
    const size_t browS = (size_t)brow * S_;
    int cur = 0;

    // Elementwise-phase mapping: each thread owns 1 batch row x 4 columns.
    const int eb  = tid >> 2;                 // batch row 0..63
    const int ej0 = (tid & 3) * 4;            // first column (0,4,8,12)
    const int emb = eb >> 4, emm = eb & 15;

    v16h ax[4];
    load_x_frags(x16, x, browS, 0, half, ax); // prefetch t=0

    for (int t = 0; t < S_; ++t) {
        // ---- 1. Issue ALL h_{t-1} fragment loads first ----
        const _Float16* hb = (cur ? hbuf1 : hbuf0) + (size_t)brow * H_;
        v16h ah[8];
        #pragma unroll
        for (int kk = 0; kk < 8; ++kk) {
            v16h a;
            #pragma unroll
            for (int p = 0; p < 8; ++p) {
                int k0 = kk * 32 + kmap_pair_base(half, p);
                h2 v = *(const h2*)(hb + k0);
                a[2 * p] = v.x; a[2 * p + 1] = v.y;
            }
            ah[kk] = a;
        }

        v8f acc0, acc1;
        #pragma unroll
        for (int e = 0; e < 8; ++e) { acc0[e] = bias[0]; acc1[e] = bias[1]; }

        // ---- 2. x_t @ U (K = 128), fragments prefetched last iteration;
        //         h-load latency hides behind these 8 WMMAs ----
        #pragma unroll
        for (int kk = 0; kk < 4; ++kk) {
            acc0 = __builtin_amdgcn_wmma_f32_16x16x32_f16(false, ax[kk], false, ufrag[0][kk],
                                                          (short)0, acc0, false, false);
            acc1 = __builtin_amdgcn_wmma_f32_16x16x32_f16(false, ax[kk], false, ufrag[1][kk],
                                                          (short)0, acc1, false, false);
        }

        // Pin all 8 h-fragments into distinct live registers here: forces the
        // 16 b128 loads to issue as one clause above (no register reuse /
        // load-use serialization) with a single s_wait before the h-WMMAs.
        asm volatile("" : "+v"(ah[0]), "+v"(ah[1]), "+v"(ah[2]), "+v"(ah[3]),
                          "+v"(ah[4]), "+v"(ah[5]), "+v"(ah[6]), "+v"(ah[7]));

        // ---- 3. h_{t-1} @ W (K = 256) ----
        #pragma unroll
        for (int kk = 0; kk < 8; ++kk) {
            acc0 = __builtin_amdgcn_wmma_f32_16x16x32_f16(false, ah[kk], false, wfrag[0][kk],
                                                          (short)0, acc0, false, false);
            acc1 = __builtin_amdgcn_wmma_f32_16x16x32_f16(false, ah[kk], false, wfrag[1][kk],
                                                          (short)0, acc1, false, false);
        }

        // ---- gate tiles -> LDS ----
        #pragma unroll
        for (int r = 0; r < 8; ++r) {
            int m = r + 8 * half;
            gLDS[g0][mb][m][n]     = acc0[r];
            gLDS[g0 + 1][mb][m][n] = acc1[r];
        }
        __syncthreads();

        // ---- elementwise gates: 1 row x 4 cols per thread, wide LDS/mem ops ----
        _Float16* hnext = (cur ? hbuf0 : hbuf1);
        {
            f4 gf = *(const f4*)&gLDS[0][emb][emm][ej0];
            f4 gi = *(const f4*)&gLDS[1][emb][emm][ej0];
            f4 go = *(const f4*)&gLDS[2][emb][emm][ej0];
            f4 gc = *(const f4*)&gLDS[3][emb][emm][ej0];
            f4 cold = *(const f4*)&cLDS[eb][ej0];
            f4 cnew, hnew;
            h4 hv;
            #pragma unroll
            for (int q = 0; q < 4; ++q) {
                float f_ = sigf(gf[q]);
                float i_ = sigf(gi[q]);
                float o_ = sigf(go[q]);
                float ct = sigf(gc[q]);     // reference: sigmoid candidate
                float c_new = f_ * cold[q] + i_ * ct;
                cnew[q] = c_new;
                float h_new = o_ * tanh_fast(c_new);
                hnew[q] = h_new;
                hv[q] = (_Float16)h_new;
            }
            *(f4*)&cLDS[eb][ej0] = cnew;
            *(h4*)(hnext + (size_t)eb * H_ + hc0 + ej0) = hv;
            __builtin_nontemporal_store(hnew, (f4*)&out[((size_t)eb * S_ + t) * H_ + hc0 + ej0]);
            if (t == S_ - 1) {
                size_t base = (size_t)B_ * S_ * H_;
                *(f4*)&out[base + (size_t)eb * H_ + hc0 + ej0] = hnew;                    // h_T
                *(f4*)&out[base + (size_t)B_ * H_ + (size_t)eb * H_ + hc0 + ej0] = cnew;  // c_T
            }
        }

        // ---- prefetch next x_t fragments; latency hides behind the barrier ----
        if (t + 1 < S_)
            load_x_frags(x16, x, browS, t + 1, half, ax);

        // ---- grid barrier across the 16 persistent WGs ----
        __threadfence();
        __syncthreads();
        if (tid == 0) {
            __hip_atomic_fetch_add(bar, 1, __ATOMIC_ACQ_REL, __HIP_MEMORY_SCOPE_AGENT);
            const int target = NWG * (t + 1);
            while (__hip_atomic_load(bar, __ATOMIC_ACQUIRE, __HIP_MEMORY_SCOPE_AGENT) < target) {
                __builtin_amdgcn_s_sleep(1);
            }
        }
        __syncthreads();
        cur ^= 1;
    }
}

extern "C" void kernel_launch(void* const* d_in, const int* in_sizes, int n_in,
                              void* d_out, int out_size, void* d_ws, size_t ws_size,
                              hipStream_t stream) {
    const float* x  = (const float*)d_in[0];
    const float* Uf = (const float*)d_in[1];
    const float* Wf = (const float*)d_in[2];
    const float* bf = (const float*)d_in[3];
    const float* Ui = (const float*)d_in[4];
    const float* Wi = (const float*)d_in[5];
    const float* bi = (const float*)d_in[6];
    const float* Uo = (const float*)d_in[7];
    const float* Wo = (const float*)d_in[8];
    const float* bo = (const float*)d_in[9];
    const float* Uc = (const float*)d_in[10];
    const float* Wc = (const float*)d_in[11];
    const float* bc = (const float*)d_in[12];
    float* out = (float*)d_out;

    char* ws = (char*)d_ws;
    int*      bar   = (int*)ws;                        // barrier counter
    _Float16* hbuf0 = (_Float16*)(ws + 256);           // [B, H] f16, ping
    _Float16* hbuf1 = hbuf0 + (size_t)B_ * H_;         // [B, H] f16, pong

    const size_t X16_OFF = 1ull << 20;                 // 1 MB
    const size_t X16_BYTES = (size_t)B_ * S_ * I_ * sizeof(_Float16);  // 32 MB
    _Float16* x16 = nullptr;
    if (ws_size >= X16_OFF + X16_BYTES) {
        x16 = (_Float16*)(ws + X16_OFF);
        const size_t nvec = (size_t)B_ * S_ * I_ / 8;  // 8 elems per thread
        xconv_kernel<<<(unsigned)(nvec / NTH), NTH, 0, stream>>>(x, x16);
    }

    lstm_init_kernel<<<(B_ * H_ + NTH - 1) / NTH, NTH, 0, stream>>>(bar, hbuf0);
    lstm_persistent_kernel<<<NWG, NTH, 0, stream>>>(
        x, x16, Uf, Wf, bf, Ui, Wi, bi, Uo, Wo, bo, Uc, Wc, bc,
        out, hbuf0, hbuf1, bar);
}
